// EfficientNetB0_unfreeze_BiGRU_MambaSSM_83580063580384
// MI455X (gfx1250) — matrix operand order repaired and checked by reading, not compiled, not run on hardware
//
#include <hip/hip_runtime.h>
#include <hip/hip_bf16.h>
#include <math.h>

// ---------------------------------------------------------------------------
// Types for CDNA5 WMMA (wave32): v_wmma_f32_16x16x32_bf16
// ---------------------------------------------------------------------------
typedef __bf16 bf16;
typedef __attribute__((ext_vector_type(16))) __bf16 bf16x16;
typedef __attribute__((ext_vector_type(8)))  __bf16 bf16x8;
typedef __attribute__((ext_vector_type(8)))  float  f32x8;

// Pointer types for the gfx1250 async global->LDS builtin (per hipcc diag:
// param0 = 'int __attribute__((__vector_size__(16))) __device__ *').
typedef int v4i_vs __attribute__((vector_size(16)));
typedef __attribute__((address_space(1))) v4i_vs* v4i_gptr;   // global (AS1)
typedef __attribute__((address_space(3))) v4i_vs* v4i_lptr;   // LDS (AS3)

#define F32X8_ZERO {0.f,0.f,0.f,0.f,0.f,0.f,0.f,0.f}

#if defined(__HIP_DEVICE_COMPILE__) && __has_builtin(__builtin_amdgcn_global_load_async_to_lds_b128)
#define HAVE_ASYNC_LDS 1
#endif

__device__ __forceinline__ f32x8 wmma_bf16(bf16x16 a, bf16x16 b, f32x8 c) {
  // 8 args: (neg_a, A, neg_b, B, c_mod, C, reuse_a, reuse_b)
  return __builtin_amdgcn_wmma_f32_16x16x32_bf16(false, a, false, b, (short)0, c,
                                                 false, false);
}

// Async (ASYNCcnt-tracked) 128-bit global -> LDS stage, with sync fallback.
__device__ __forceinline__ void stage_b128(const bf16* gsrc, bf16* ldst) {
#ifdef HAVE_ASYNC_LDS
  __builtin_amdgcn_global_load_async_to_lds_b128(
      (v4i_gptr)gsrc, (v4i_lptr)ldst, 0, 0);
#else
  *(bf16x8*)ldst = *(const bf16x8*)gsrc;
#endif
}

__device__ __forceinline__ void wait_async0() {
#ifdef HAVE_ASYNC_LDS
#if __has_builtin(__builtin_amdgcn_s_wait_asynccnt)
  __builtin_amdgcn_s_wait_asynccnt(0);
#else
  asm volatile("s_wait_asynccnt 0x0" ::: "memory");
#endif
#endif
}

// A fragment: 16x32 bf16 tile, row-major source with leading dim `lda`.
// Contiguous-K layout used here (same as load_frag_b): lanes 0-15 -> M=lane, VGPR0..3 = K[half*16 .. half*16+8),
//             VGPR4..7 = K[half*16+8 .. half*16+16), half = lane>>4.
__device__ __forceinline__ bf16x16 load_frag_a(const bf16* __restrict__ base,
                                               int lda, int lane) {
  const int m    = lane & 15;
  const int half = lane >> 4;
  const bf16* p0 = base + (size_t)m * lda + half * 16;
  const bf16* p1 = p0 + 8;
  union { bf16x16 v; bf16x8 h[2]; } u;
  u.h[0] = *(const bf16x8*)p0;
  u.h[1] = *(const bf16x8*)p1;
  return u.v;
}

// B fragment: B = W^T where W is (N x K) row-major with leading dim `ldw`.
// Lane holds column n = lane&15; lanes 0-15 carry K[0..16), lanes 16-31 K[16..32),
// packed 2 per VGPR -> 16 contiguous K values per lane.
__device__ __forceinline__ bf16x16 load_frag_b(const bf16* __restrict__ wbase,
                                               int ldw, int lane) {
  const int n    = lane & 15;
  const int half = lane >> 4;
  const bf16* p  = wbase + (size_t)n * ldw + half * 16;
  return *(const bf16x16*)p;
}

// ---------------------------------------------------------------------------
// Generic bf16 WMMA GEMM: out[M,N] = act(A[M,K] @ W[N,K]^T + bias[N])
// Block: 256 threads = 8 waves arranged 4(M) x 2(N); each wave does a 16x32
// strip of C (2 accumulators sharing the A fragment). Block tile = 64x64.
// A block tile (64x32) is double-buffered through LDS via async-to-LDS b128
// copies (ASYNCcnt), overlapping the next tile's fill with current WMMAs.
// ---------------------------------------------------------------------------
template <typename OutT, int ACT>
__global__ void __launch_bounds__(256) gemm_bf16_kernel(
    const bf16* __restrict__ A, const bf16* __restrict__ W,
    const float* __restrict__ bias, OutT* __restrict__ out,
    int M, int N, int K) {
  __shared__ bf16 Asmem[2][64][32];

  const int tid  = threadIdx.x;
  const int lane = tid & 31;
  const int wave = tid >> 5;
  const int wm   = wave >> 1;                 // 0..3
  const int mblk = blockIdx.y * 64;
  const int n0   = blockIdx.x * 64 + (wave & 1) * 32;

  // Each thread stages one 16B segment of the 64x32 A tile (256 x b128).
  const int arow = tid >> 2;                  // 0..63
  const int aseg = (tid & 3) * 8;             // 0,8,16,24 (bf16 elems)

  const bf16* w0 = W + (size_t)n0 * K;
  const bf16* w1 = W + (size_t)(n0 + 16) * K;

  f32x8 acc0 = F32X8_ZERO;
  f32x8 acc1 = F32X8_ZERO;

  const int nk = K >> 5;
  // prologue: fill buffer 0
  stage_b128(A + (size_t)(mblk + arow) * K + aseg, &Asmem[0][arow][aseg]);

  for (int i = 0; i < nk; ++i) {
    wait_async0();          // buffer (i&1) fully landed in LDS
    __syncthreads();        // ...and visible to (and prior reads done by) all waves
    if (i + 1 < nk) {
      const int kn = (i + 1) << 5;
      stage_b128(A + (size_t)(mblk + arow) * K + kn + aseg,
                 &Asmem[(i + 1) & 1][arow][aseg]);
    }
    __builtin_prefetch(w0 + (i << 5) + 256, 0, 0);  // global_prefetch_b8
    bf16x16 a  = load_frag_a(&Asmem[i & 1][wm * 16][0], 32, lane);
    bf16x16 b0 = load_frag_b(w0 + (i << 5), K, lane);
    bf16x16 b1 = load_frag_b(w1 + (i << 5), K, lane);
    acc0 = wmma_bf16(a, b0, acc0);
    acc1 = wmma_bf16(a, b1, acc1);
  }

  const int half = lane >> 4;
  const int n    = lane & 15;
  const float bb0 = bias ? bias[n0 + n] : 0.f;
  const float bb1 = bias ? bias[n0 + 16 + n] : 0.f;
#pragma unroll
  for (int r = 0; r < 8; ++r) {
    const int m = mblk + wm * 16 + r + 8 * half;
    float v0 = acc0[r] + bb0;
    float v1 = acc1[r] + bb1;
    if (ACT == 1) { v0 = fmaxf(v0, 0.f); v1 = fmaxf(v1, 0.f); }
    out[(size_t)m * N + (n0 + n)]      = (OutT)v0;
    out[(size_t)m * N + (n0 + 16 + n)] = (OutT)v1;
  }
}

// ---------------------------------------------------------------------------
// Prep: fold residual+proj weights and eval-mode BN into one bf16 weight + bias
// W'[d,f] = (Wr+Wp)[d,f] * s_d ; b'[d] = (br+bp-bn_m)[d]*s_d + bn_b[d]
// s_d = bn_g[d] * rsqrt(bn_v[d]+eps)
// ---------------------------------------------------------------------------
__global__ void prep_backbone_kernel(
    const float* __restrict__ Wr, const float* __restrict__ br,
    const float* __restrict__ Wp, const float* __restrict__ bp,
    const float* __restrict__ bn_g, const float* __restrict__ bn_b,
    const float* __restrict__ bn_m, const float* __restrict__ bn_v,
    bf16* __restrict__ Wc, float* __restrict__ bc, int D, int F) {
  const int i = blockIdx.x * blockDim.x + threadIdx.x;
  if (i >= D * F) return;
  const int d = i / F;
  const int f = i - d * F;
  const float s = bn_g[d] * rsqrtf(bn_v[d] + 1e-5f);
  Wc[i] = (bf16)((Wr[i] + Wp[i]) * s);
  if (f == 0) bc[d] = (br[d] + bp[d] - bn_m[d]) * s + bn_b[d];
}

__global__ void cvt_bf16_kernel(const float* __restrict__ src,
                                bf16* __restrict__ dst, int n) {
  const int i = blockIdx.x * blockDim.x + threadIdx.x;
  if (i < n) dst[i] = (bf16)src[i];
}

// ---------------------------------------------------------------------------
// Persistent BiGRU scan. grid.x = 16: (8 batch-tiles of 16) x (2 directions).
// Recurrent state h lives in LDS (f32 master + bf16 copy for WMMA).
// Per step: gh = h @ Whh^T via WMMA (24 N-tiles over 3H=384, 3 per wave),
// then elementwise PyTorch GRU gates using precomputed gi (bf16, global).
// ---------------------------------------------------------------------------
__global__ void __launch_bounds__(256) gru_scan_kernel(
    const bf16* __restrict__ WhhF, const bf16* __restrict__ WhhB,
    const float* __restrict__ bhhF, const float* __restrict__ bhhB,
    const bf16* __restrict__ giF, const bf16* __restrict__ giB,
    bf16* __restrict__ g_out) {
  const int dir = blockIdx.x & 1;
  const int b0  = (blockIdx.x >> 1) * 16;
  const bf16*  Whh = dir ? WhhB : WhhF;
  const float* bhh = dir ? bhhB : bhhF;
  const bf16*  gi  = dir ? giB  : giF;

  __shared__ float hF[16][128];
  __shared__ bf16  hB[16][128];
  __shared__ float gh[16][384];

  const int tid  = threadIdx.x;
  const int lane = tid & 31;
  const int wave = tid >> 5;

  for (int i = tid; i < 16 * 128; i += 256) {
    hF[i >> 7][i & 127] = 0.f;
    hB[i >> 7][i & 127] = (bf16)0.f;
  }
  __syncthreads();

  for (int t = 0; t < 128; ++t) {
    const int s = dir ? (127 - t) : t;

    // ---- gh = hB @ Whh^T + bhh -------------------------------------------
#pragma unroll
    for (int i = 0; i < 3; ++i) {
      const int n0 = (wave * 3 + i) * 16;
      f32x8 acc = F32X8_ZERO;
#pragma unroll
      for (int k0 = 0; k0 < 128; k0 += 32) {
        bf16x16 a = load_frag_a(&hB[0][0] + k0, 128, lane);
        bf16x16 b = load_frag_b(Whh + (size_t)n0 * 128 + k0, 128, lane);
        acc = wmma_bf16(a, b, acc);
      }
      const int half = lane >> 4;
      const int n    = lane & 15;
      const float bb = bhh[n0 + n];
#pragma unroll
      for (int r = 0; r < 8; ++r) gh[r + 8 * half][n0 + n] = acc[r] + bb;
    }
    __syncthreads();

    // ---- gates ------------------------------------------------------------
    const int row     = tid >> 4;
    const int colbase = (tid & 15) * 8;
    const int b       = b0 + row;
    const bf16* gir   = gi + ((size_t)b * 128 + s) * 384;
#pragma unroll
    for (int j = 0; j < 8; ++j) {
      const int hc = colbase + j;
      const float ir  = (float)gir[hc];
      const float iz  = (float)gir[hc + 128];
      const float in_ = (float)gir[hc + 256];
      const float hr = gh[row][hc];
      const float hz = gh[row][hc + 128];
      const float hn = gh[row][hc + 256];
      const float r_ = 1.f / (1.f + __expf(-(ir + hr)));
      const float z_ = 1.f / (1.f + __expf(-(iz + hz)));
      const float nc = tanhf(in_ + r_ * hn);
      const float hp = hF[row][hc];
      const float hv = (1.f - z_) * nc + z_ * hp;
      hF[row][hc] = hv;
      hB[row][hc] = (bf16)hv;
      g_out[((size_t)b * 128 + s) * 256 + dir * 128 + hc] = (bf16)hv;
    }
    __syncthreads();
  }
}

// ---------------------------------------------------------------------------
// Persistent Mamba scan: h_t = GELU(h_{t-1} @ A^T + Bu_t); hs streamed to
// global (bf16) for the y = hs @ C^T GEMM. grid.x = 8 batch tiles of 16.
// 16 N-tiles over M=256, 2 per wave; K=256 -> 8 WMMA per tile per step.
// ---------------------------------------------------------------------------
__global__ void __launch_bounds__(256) mamba_scan_kernel(
    const bf16* __restrict__ A_bf, const bf16* __restrict__ bu,
    bf16* __restrict__ hs) {
  const int b0   = blockIdx.x * 16;
  const int tid  = threadIdx.x;
  const int lane = tid & 31;
  const int wave = tid >> 5;

  __shared__ bf16 hB[16][256];
  for (int i = tid; i < 16 * 256; i += 256) hB[i >> 8][i & 255] = (bf16)0.f;
  __syncthreads();

  const int n0a = (wave * 2 + 0) * 16;
  const int n0b = (wave * 2 + 1) * 16;

  for (int t = 0; t < 128; ++t) {
    f32x8 acc0 = F32X8_ZERO;
    f32x8 acc1 = F32X8_ZERO;
#pragma unroll
    for (int k0 = 0; k0 < 256; k0 += 32) {
      bf16x16 a  = load_frag_a(&hB[0][0] + k0, 256, lane);
      bf16x16 w0 = load_frag_b(A_bf + (size_t)n0a * 256 + k0, 256, lane);
      bf16x16 w1 = load_frag_b(A_bf + (size_t)n0b * 256 + k0, 256, lane);
      acc0 = wmma_bf16(a, w0, acc0);
      acc1 = wmma_bf16(a, w1, acc1);
    }
    __syncthreads();  // all reads of hB done

    const int half = lane >> 4;
    const int n    = lane & 15;
#pragma unroll
    for (int r = 0; r < 8; ++r) {
      const int m = r + 8 * half;
      const int b = b0 + m;
      const size_t rowoff = ((size_t)b * 128 + t) * 256;
      float v0 = acc0[r] + (float)bu[rowoff + n0a + n];
      float v1 = acc1[r] + (float)bu[rowoff + n0b + n];
      v0 = 0.5f * v0 * (1.f + erff(v0 * 0.70710678118f));   // exact GELU
      v1 = 0.5f * v1 * (1.f + erff(v1 * 0.70710678118f));
      hB[m][n0a + n] = (bf16)v0;
      hB[m][n0b + n] = (bf16)v1;
      hs[rowoff + n0a + n] = (bf16)v0;
      hs[rowoff + n0b + n] = (bf16)v1;
    }
    __syncthreads();
  }
}

// ---------------------------------------------------------------------------
// LayerNorm(256) + FC(256->2). One block per (b,s) row.
// ---------------------------------------------------------------------------
__global__ void __launch_bounds__(256) ln_fc_kernel(
    const float* __restrict__ y, const float* __restrict__ ln_g,
    const float* __restrict__ ln_b, const float* __restrict__ Wfc,
    const float* __restrict__ bfc, float* __restrict__ out) {
  const int row = blockIdx.x;
  const int tid = threadIdx.x;
  __shared__ float red[256];

  const float v = y[(size_t)row * 256 + tid];
  red[tid] = v; __syncthreads();
#pragma unroll
  for (int s = 128; s > 0; s >>= 1) {
    if (tid < s) red[tid] += red[tid + s];
    __syncthreads();
  }
  const float mu = red[0] * (1.f / 256.f);
  __syncthreads();

  const float d = v - mu;
  red[tid] = d * d; __syncthreads();
#pragma unroll
  for (int s = 128; s > 0; s >>= 1) {
    if (tid < s) red[tid] += red[tid + s];
    __syncthreads();
  }
  const float var = red[0] * (1.f / 256.f);
  __syncthreads();

  const float yn = d * rsqrtf(var + 1e-5f) * ln_g[tid] + ln_b[tid];

  red[tid] = yn * Wfc[tid]; __syncthreads();
#pragma unroll
  for (int s = 128; s > 0; s >>= 1) {
    if (tid < s) red[tid] += red[tid + s];
    __syncthreads();
  }
  if (tid == 0) out[(size_t)row * 2 + 0] = red[0] + bfc[0];
  __syncthreads();

  red[tid] = yn * Wfc[256 + tid]; __syncthreads();
#pragma unroll
  for (int s = 128; s > 0; s >>= 1) {
    if (tid < s) red[tid] += red[tid + s];
    __syncthreads();
  }
  if (tid == 0) out[(size_t)row * 2 + 1] = red[0] + bfc[1];
}

// ---------------------------------------------------------------------------
// Host orchestration
// ---------------------------------------------------------------------------
extern "C" void kernel_launch(void* const* d_in, const int* in_sizes, int n_in,
                              void* d_out, int out_size, void* d_ws, size_t ws_size,
                              hipStream_t stream) {
  (void)in_sizes; (void)n_in; (void)out_size; (void)ws_size;

  constexpr int B = 128, S = 128, F = 1280, D = 256, H = 128, M = 256;
  constexpr int R = B * S;            // 16384 sequence rows
  constexpr int G3 = 3 * H;           // 384

  const float* x     = (const float*)d_in[0];
  const float* Wr    = (const float*)d_in[1];
  const float* br    = (const float*)d_in[2];
  const float* Wp    = (const float*)d_in[3];
  const float* bp    = (const float*)d_in[4];
  const float* bn_g  = (const float*)d_in[5];
  const float* bn_b  = (const float*)d_in[6];
  const float* bn_m  = (const float*)d_in[7];
  const float* bn_v  = (const float*)d_in[8];
  const float* Wih_f = (const float*)d_in[9];
  const float* Whh_f = (const float*)d_in[10];
  const float* bih_f = (const float*)d_in[11];
  const float* bhh_f = (const float*)d_in[12];
  const float* Wih_b = (const float*)d_in[13];
  const float* Whh_b = (const float*)d_in[14];
  const float* bih_b = (const float*)d_in[15];
  const float* bhh_b = (const float*)d_in[16];
  const float* A     = (const float*)d_in[17];
  const float* Bm    = (const float*)d_in[18];
  const float* C     = (const float*)d_in[19];
  const float* ln_g  = (const float*)d_in[20];
  const float* ln_b  = (const float*)d_in[21];
  const float* Wfc   = (const float*)d_in[22];
  const float* bfc   = (const float*)d_in[23];
  float* out = (float*)d_out;

  // --- carve workspace (256B aligned slices) ---
  char* ws = (char*)d_ws;
  size_t off = 0;
  auto take = [&](size_t bytes) -> char* {
    char* p = ws + off;
    off += (bytes + 255) & ~(size_t)255;
    return p;
  };
  bf16*  x_bf   = (bf16*)take((size_t)R * F * 2);
  bf16*  Wc_bf  = (bf16*)take((size_t)D * F * 2);
  float* bc     = (float*)take((size_t)D * 4);
  bf16*  WihF   = (bf16*)take((size_t)G3 * D * 2);
  bf16*  WihB   = (bf16*)take((size_t)G3 * D * 2);
  bf16*  WhhF   = (bf16*)take((size_t)G3 * H * 2);
  bf16*  WhhB   = (bf16*)take((size_t)G3 * H * 2);
  bf16*  A_bf   = (bf16*)take((size_t)M * M * 2);
  bf16*  Bm_bf  = (bf16*)take((size_t)M * (2 * H) * 2);
  bf16*  C_bf   = (bf16*)take((size_t)M * M * 2);
  bf16*  feat   = (bf16*)take((size_t)R * D * 2);
  bf16*  giF    = (bf16*)take((size_t)R * G3 * 2);
  bf16*  giB    = (bf16*)take((size_t)R * G3 * 2);
  bf16*  g_bf   = (bf16*)take((size_t)R * (2 * H) * 2);
  bf16*  bu_bf  = (bf16*)take((size_t)R * M * 2);
  bf16*  hs_bf  = (bf16*)take((size_t)R * M * 2);
  float* y_f32  = (float*)take((size_t)R * M * 4);

  auto cvt = [&](const float* src, bf16* dst, int n) {
    cvt_bf16_kernel<<<(n + 255) / 256, 256, 0, stream>>>(src, dst, n);
  };

  // 1) weight prep / precision conversion
  prep_backbone_kernel<<<(D * F + 255) / 256, 256, 0, stream>>>(
      Wr, br, Wp, bp, bn_g, bn_b, bn_m, bn_v, Wc_bf, bc, D, F);
  cvt(x, x_bf, R * F);
  cvt(Wih_f, WihF, G3 * D);
  cvt(Wih_b, WihB, G3 * D);
  cvt(Whh_f, WhhF, G3 * H);
  cvt(Whh_b, WhhB, G3 * H);
  cvt(A, A_bf, M * M);
  cvt(Bm, Bm_bf, M * 2 * H);
  cvt(C, C_bf, M * M);

  // 2) fused backbone head: feat = relu(BN(x @ (Wr+Wp)^T + br+bp))
  gemm_bf16_kernel<bf16, 1><<<dim3(D / 64, R / 64), 256, 0, stream>>>(
      x_bf, Wc_bf, bc, feat, R, D, F);

  // 3) GRU input gates (both directions): gi = feat @ Wih^T + bih
  gemm_bf16_kernel<bf16, 0><<<dim3(G3 / 64, R / 64), 256, 0, stream>>>(
      feat, WihF, bih_f, giF, R, G3, D);
  gemm_bf16_kernel<bf16, 0><<<dim3(G3 / 64, R / 64), 256, 0, stream>>>(
      feat, WihB, bih_b, giB, R, G3, D);

  // 4) persistent BiGRU scan (fwd+bwd interleaved across blocks)
  gru_scan_kernel<<<16, 256, 0, stream>>>(WhhF, WhhB, bhh_f, bhh_b, giF, giB, g_bf);

  // 5) Mamba input projection: bu = g @ Bm^T
  gemm_bf16_kernel<bf16, 0><<<dim3(M / 64, R / 64), 256, 0, stream>>>(
      g_bf, Bm_bf, (const float*)nullptr, bu_bf, R, M, 2 * H);

  // 6) persistent Mamba scan: h_t = gelu(h @ A^T + bu_t)
  mamba_scan_kernel<<<8, 256, 0, stream>>>(A_bf, bu_bf, hs_bf);

  // 7) y = hs @ C^T (f32 out for LayerNorm precision)
  gemm_bf16_kernel<float, 0><<<dim3(M / 64, R / 64), 256, 0, stream>>>(
      hs_bf, C_bf, (const float*)nullptr, y_f32, R, M, M);

  // 8) LayerNorm(256) + FC(256->2)
  ln_fc_kernel<<<R, 256, 0, stream>>>(y_f32, ln_g, ln_b, Wfc, bfc, out);
}
